// Net_25520695673338
// MI455X (gfx1250) — matrix operand order repaired and checked by reading, not compiled, not run on hardware
//
#include <hip/hip_runtime.h>
#include <hip/hip_bf16.h>

// ---------------- problem constants (match reference) ----------------
#define BGRAPH 32768
#define NPG    54
#define EPG    144
#define NTOT   (BGRAPH * NPG)       // 1,769,472
#define ETOT   (BGRAPH * EPG)       // 4,718,592
#define GLOB   10
#define F0     8
#define F1     16
#define F2     4
#define KPAD   256                  // 226 padded up for WMMA K
#define NH     128                  // out-MLP hidden

typedef __attribute__((ext_vector_type(16))) _Float16 v16h;
typedef __attribute__((ext_vector_type(8)))  _Float16 v8h;
typedef __attribute__((ext_vector_type(8)))  float    v8f;

// ---------------------------------------------------------------------
// Kernel 1: pack Wo1 [226,128] f32 -> WMMA B-fragment layout, f16, K padded
// Bp[ntile][kstep][lane][i] ; lane c = lane&15, h = lane>>4 :
//   b[i] = W[kk + h*16 + i][ntile*16 + c]   (contiguous k order, same as the A loads in out_mlp)
// ---------------------------------------------------------------------
__global__ __launch_bounds__(256) void pack_w1(const float* __restrict__ Wo1,
                                               _Float16* __restrict__ Bp) {
    int idx = blockIdx.x * 256 + threadIdx.x;      // 8*8*32*16 = 32768 total
    if (idx >= 8 * 8 * 32 * 16) return;
    int i     = idx & 15;
    int lane  = (idx >> 4) & 31;
    int ks    = (idx >> 9) & 7;
    int ntile = idx >> 12;
    int k = ks * 32 + (lane >> 4) * 16 + i;
    int n = ntile * 16 + (lane & 15);
    _Float16 v = (_Float16)0.f;
    if (k < NPG * F2 + GLOB) v = (_Float16)Wo1[k * NH + n];
    Bp[idx] = v;
}

// ---------------------------------------------------------------------
// Kernel 2: fully fused per-graph GNN (both GraphConvs stay in LDS).
// One 64-thread block (2 waves) per graph. Writes relu(h2) as f16 into
// the padded activation matrix X16[B, 256], cols 0..215.
// ---------------------------------------------------------------------
__global__ __launch_bounds__(64) void gnn_fused(
    const float* __restrict__ x, const int* __restrict__ ei,
    const float* __restrict__ ea,
    const float* __restrict__ Wr1, const float* __restrict__ b1,
    const float* __restrict__ Wroot1,
    const float* __restrict__ Wr2, const float* __restrict__ b2,
    const float* __restrict__ Wroot2,
    _Float16* __restrict__ X16) {

    __shared__ float xs [NPG * F0];     // 432
    __shared__ float h1 [NPG * F1];     // 864
    __shared__ float agg[NPG * F1];     // 864 (reused: conv1 uses first 432)
    __shared__ float wr1[F0 * F1], wo1[F0 * F1], wb1[F1];
    __shared__ float wr2[F1 * F2], wo2[F1 * F2], wb2[F2];

    const int t    = threadIdx.x;
    const int g    = blockIdx.x;
    const int base = g * NPG;

    // cooperative weight load (broadcast-friendly, L2 resident)
    for (int i = t; i < F0 * F1; i += 64) { wr1[i] = Wr1[i]; wo1[i] = Wroot1[i]; }
    for (int i = t; i < F1 * F2; i += 64) { wr2[i] = Wr2[i]; wo2[i] = Wroot2[i]; }
    if (t < F1) wb1[t] = b1[t];
    if (t < F2) wb2[t] = b2[t];

    // load node features, zero conv1 accumulator
    for (int i = t; i < NPG * F0; i += 64) { xs[i] = x[base * F0 + i]; agg[i] = 0.f; }
    __syncthreads();

    // ---- conv1 message+aggregate: agg[dst] += w * x[src] (LDS atomics) ----
    for (int e = t; e < EPG; e += 64) {
        int   s = ei[g * EPG + e] - base;
        int   d = ei[(size_t)ETOT + g * EPG + e] - base;
        float w = ea[g * EPG + e];
        #pragma unroll
        for (int f = 0; f < F0; ++f)
            atomicAdd(&agg[d * F0 + f], w * xs[s * F0 + f]);
    }
    __syncthreads();

    // ---- h1 = relu(agg@Wrel1 + b1 + x@Wroot1) ----
    if (t < NPG) {
        #pragma unroll
        for (int j = 0; j < F1; ++j) {
            float acc = wb1[j];
            #pragma unroll
            for (int k = 0; k < F0; ++k)
                acc += agg[t * F0 + k] * wr1[k * F1 + j]
                     +  xs[t * F0 + k] * wo1[k * F1 + j];
            h1[t * F1 + j] = acc > 0.f ? acc : 0.f;
        }
    }
    __syncthreads();
    for (int i = t; i < NPG * F1; i += 64) agg[i] = 0.f;
    __syncthreads();

    // ---- conv2 message+aggregate on h1 ----
    for (int e = t; e < EPG; e += 64) {
        int   s = ei[g * EPG + e] - base;
        int   d = ei[(size_t)ETOT + g * EPG + e] - base;
        float w = ea[g * EPG + e];
        #pragma unroll
        for (int f = 0; f < F1; ++f)
            atomicAdd(&agg[d * F1 + f], w * h1[s * F1 + f]);
    }
    __syncthreads();

    // ---- h2 = relu(agg@Wrel2 + b2 + h1@Wroot2) -> f16 embeds ----
    if (t < NPG) {
        #pragma unroll
        for (int j = 0; j < F2; ++j) {
            float acc = wb2[j];
            #pragma unroll
            for (int k = 0; k < F1; ++k)
                acc += agg[t * F1 + k] * wr2[k * F2 + j]
                     +  h1[t * F1 + k] * wo2[k * F2 + j];
            acc = acc > 0.f ? acc : 0.f;
            X16[(size_t)g * KPAD + t * F2 + j] = (_Float16)acc;
        }
    }
}

// ---------------------------------------------------------------------
// Kernel 3: global-feature MLP 10->8->8->10, one thread per graph.
// Writes cols 216..225 of X16 and zero-pads cols 226..255.
// ---------------------------------------------------------------------
__global__ __launch_bounds__(256) void global_mlp(
    const float* __restrict__ gf,
    const float* __restrict__ Wg1, const float* __restrict__ bg1,
    const float* __restrict__ Wg2, const float* __restrict__ bg2,
    const float* __restrict__ Wg3, const float* __restrict__ bg3,
    _Float16* __restrict__ X16) {

    int g = blockIdx.x * 256 + threadIdx.x;
    if (g >= BGRAPH) return;
    float in[GLOB], a1[8], a2[8];
    #pragma unroll
    for (int k = 0; k < GLOB; ++k) in[k] = gf[g * GLOB + k];
    #pragma unroll
    for (int j = 0; j < 8; ++j) {
        float acc = bg1[j];
        #pragma unroll
        for (int k = 0; k < GLOB; ++k) acc += in[k] * Wg1[k * 8 + j];
        a1[j] = acc > 0.f ? acc : 0.f;
    }
    #pragma unroll
    for (int j = 0; j < 8; ++j) {
        float acc = bg2[j];
        #pragma unroll
        for (int k = 0; k < 8; ++k) acc += a1[k] * Wg2[k * 8 + j];
        a2[j] = acc > 0.f ? acc : 0.f;
    }
    _Float16* row = X16 + (size_t)g * KPAD + NPG * F2;   // col 216
    #pragma unroll
    for (int j = 0; j < GLOB; ++j) {
        float acc = bg3[j];
        #pragma unroll
        for (int k = 0; k < 8; ++k) acc += a2[k] * Wg3[k * GLOB + j];
        acc = acc > 0.f ? acc : 0.f;
        row[j] = (_Float16)acc;
    }
    #pragma unroll
    for (int j = GLOB; j < KPAD - NPG * F2; ++j) row[j] = (_Float16)0.f; // pad
}

// ---------------------------------------------------------------------
// Kernel 4: out-MLP with WMMA.
//   Y = relu(X16[32768,256] @ W1 + bo1)  via v_wmma_f32_16x16x32_f16
//   out = sigmoid(Y @ Wo2 + bo2)
// Block = 256 threads (8 waves); each block owns a 16-row M tile; wave w
// owns N tile w (8*16 = 128). A frags from row-major X16 in contiguous
// k order (k = 16h + s); B frags from pre-packed Bp (1 contiguous 32B load / lane).
// ---------------------------------------------------------------------
__global__ __launch_bounds__(256) void out_mlp(
    const _Float16* __restrict__ X16, const _Float16* __restrict__ Bp,
    const float* __restrict__ bo1, const float* __restrict__ Wo2,
    const float* __restrict__ bo2, float* __restrict__ out) {

    __shared__ float Yt[16][NH + 4];            // relu'd hidden tile, padded

    const int lane = threadIdx.x & 31;
    const int wave = threadIdx.x >> 5;          // ntile 0..7
    const int r    = lane & 15;
    const int h    = lane >> 4;
    const int m0   = blockIdx.x * 16;

    const _Float16* pA = X16 + (size_t)(m0 + r) * KPAD;
    const _Float16* pB = Bp + ((size_t)wave * 8 * 32) * 16 + (size_t)lane * 16;

    v8f c = {};
    #pragma unroll
    for (int ks = 0; ks < 8; ++ks) {
        const int kk = ks * 32;
        // A 16x32 f16 fragment: lane(r,h): K = kk+h*16..+15
        v8h a0 = *(const v8h*)(pA + kk + h * 16);
        v8h a1 = *(const v8h*)(pA + kk + h * 16 + 8);
        v16h a = __builtin_shufflevector(a0, a1, 0, 1, 2, 3, 4, 5, 6, 7,
                                         8, 9, 10, 11, 12, 13, 14, 15);
        // B 32x16 f16 fragment: pre-packed, contiguous per lane
        v16h b = *(const v16h*)(pB + (size_t)ks * 32 * 16);
        c = __builtin_amdgcn_wmma_f32_16x16x32_f16(
                false, a, false, b, (short)0, c, false, false);
    }

    // bias + relu, spill 16x16 tile into the block's 16x128 LDS matrix.
    // c[v] is row (v + h*8), col (wave*16 + r)  [ISA 32-bit C/D layout]
    const float bias = bo1[wave * 16 + r];
    #pragma unroll
    for (int v = 0; v < 8; ++v) {
        float y = c[v] + bias;
        Yt[v + h * 8][wave * 16 + r] = y > 0.f ? y : 0.f;
    }
    __syncthreads();

    // second layer: 128 -> 1, then sigmoid
    if (threadIdx.x < 16) {
        float acc = bo2[0];
        #pragma unroll 8
        for (int n = 0; n < NH; ++n) acc += Yt[threadIdx.x][n] * Wo2[n];
        out[m0 + threadIdx.x] = 1.f / (1.f + __expf(-acc));
    }
}

// ---------------------------------------------------------------------
extern "C" void kernel_launch(void* const* d_in, const int* in_sizes, int n_in,
                              void* d_out, int out_size, void* d_ws, size_t ws_size,
                              hipStream_t stream) {
    const float* x      = (const float*)d_in[0];
    const int*   ei     = (const int*)  d_in[1];
    const float* ea     = (const float*)d_in[2];
    const float* gfeat  = (const float*)d_in[3];
    const float* Wr1    = (const float*)d_in[4];
    const float* b1     = (const float*)d_in[5];
    const float* Wroot1 = (const float*)d_in[6];
    const float* Wr2    = (const float*)d_in[7];
    const float* b2     = (const float*)d_in[8];
    const float* Wroot2 = (const float*)d_in[9];
    const float* Wg1    = (const float*)d_in[10];
    const float* bg1    = (const float*)d_in[11];
    const float* Wg2    = (const float*)d_in[12];
    const float* bg2    = (const float*)d_in[13];
    const float* Wg3    = (const float*)d_in[14];
    const float* bg3    = (const float*)d_in[15];
    const float* Wo1    = (const float*)d_in[16];
    const float* bo1    = (const float*)d_in[17];
    const float* Wo2    = (const float*)d_in[18];
    const float* bo2    = (const float*)d_in[19];
    float* out = (float*)d_out;

    // workspace: padded f16 activations [B,256] (16 MiB) + packed W1 (64 KiB)
    _Float16* X16 = (_Float16*)d_ws;
    _Float16* Bp  = X16 + (size_t)BGRAPH * KPAD;

    pack_w1   <<<128, 256, 0, stream>>>(Wo1, Bp);
    gnn_fused <<<BGRAPH, 64, 0, stream>>>(x, ei, ea, Wr1, b1, Wroot1,
                                          Wr2, b2, Wroot2, X16);
    global_mlp<<<BGRAPH / 256, 256, 0, stream>>>(gfeat, Wg1, bg1, Wg2, bg2,
                                                 Wg3, bg3, X16);
    out_mlp   <<<BGRAPH / 16, 256, 0, stream>>>(X16, Bp, bo1, Wo2, bo2, out);
}
